// BackboneRaindrop_63711544869452
// MI455X (gfx1250) — compile-verified, matrix operand order checked by reading, not run on hardware
//
#include <hip/hip_runtime.h>
#include <hip/hip_bf16.h>
#include <math.h>

// ---------------------------------------------------------------------------
// Types for CDNA5 WMMA (wave32): v_wmma_f32_16x16x32_bf16
//   A: 16x32 bf16 (v16bf), B: 32x16 bf16 (v16bf), C/D: 16x16 f32 (v8f)
// ---------------------------------------------------------------------------
typedef __bf16 bf16;
typedef __bf16 v16bf __attribute__((ext_vector_type(16)));
typedef __bf16 bfx8  __attribute__((ext_vector_type(8)));
typedef float  v8f   __attribute__((ext_vector_type(8)));

// ------------------------- static model config ------------------------------
#define B_    64
#define F_    128
#define L_    256
#define DOB   8
#define DPE   16
#define NH    16
#define HD    65            // head dim = 1040/16
#define NL    2
#define DFFN  2048
#define CC    2048          // L_*DOB  (obs-prop channels)
#define DD    1040          // F_*DOB + DPE (transformer width)
#define QKVW  3120          // 3*DD
#define MBF   8192          // B_*F_   (obs-prop GEMM rows)
#define MLB   16384         // L_*B_   (transformer GEMM rows)
#define LN_EPS 1e-5f

// ===========================================================================
// Generic bf16 WMMA GEMM:  C[M,N] = act(A[M,K] @ W[N,K]^T + bias)
//   A row-major [M,K] bf16, W row-major [N,K] bf16 (both contiguous in K,
//   matching einsum('mk,nk->mn')).  fp32 accumulate in v8f.
// Block: 256 threads = 8 waves. Block tile 256x64, K-step 32.
//   waves laid out 4(m) x 2(n); each wave owns 4x2 16x16 WMMA tiles
//   (8 v_wmma per K-step vs 12 fragment ds_loads).
// Tile staging: CDNA5 GLOBAL_LOAD_ASYNC_TO_LDS_B128 (ASYNCcnt, no VGPR data
//   round-trip) on the block-uniform fast path, DOUBLE-BUFFERED so the next
//   K-tile's HBM/L2 latency overlaps the current tile's 8 WMMAs; guarded
//   scalar LDS stores on K-tail / N-edge tiles (same pipeline slot).
// Fragment layouts per CDNA5 ISA 7.12.2:
//   A 16x32 bf16: lane = m%16 + 16*kh ; elems 0..7 -> K = kh*8 + j,
//                 elems 8..15 -> K = 16 + kh*8 + (j-8)   (two 16B chunks)
//   B symmetric (rows of W play the 16-wide N role)
//   C/D 16x16 f32: vgpr g, lane l : M = g + 8*(l>=16), N = l%16
// M must be a multiple of 256 (true for 8192 / 16384); N,K fully guarded.
// ===========================================================================
#define TM 256
#define TN 64
#define TK 32

__global__ __launch_bounds__(256)
void gemm_bf16_kernel(const bf16* __restrict__ A, const bf16* __restrict__ W,
                      const float* __restrict__ bias,
                      float* __restrict__ Cf, bf16* __restrict__ Cb,
                      int M, int N, int K, int relu)
{
    __shared__ bf16 As[2][TM * TK];   // 2 x 16 KB
    __shared__ bf16 Bs[2][TN * TK];   // 2 x  4 KB

    const int tid  = threadIdx.x;
    const int lane = tid & 31;
    const int wave = tid >> 5;          // 0..7
    const int wm   = wave & 3;          // 4 m-groups of 64 rows (4 tiles each)
    const int wn   = wave >> 2;         // 2 n-groups of 32 cols (2 tiles each)
    const int lm   = lane & 15;
    const int kh   = lane >> 4;         // 0/1 (K-half selector & +8 M rows in C)

    const int m0 = blockIdx.y * TM;
    const int n0 = blockIdx.x * TN;

    v8f acc[4][2] = {};                 // zero-init accumulators (64 VGPRs)

    const int brow = tid >> 2, bseg = tid & 3;    // B stage: 64 rows x 4 segs

    // Stage one 256x32 A tile + 64x32 B tile into LDS buffer `buf`.
    // Returns true if async-to-LDS was used (needs s_wait_asynccnt).
    auto stage = [&](int k0, int buf) -> bool {
        const bool fullk = (k0 + TK) <= K;            // block-uniform
        const bool fastB = fullk && (n0 + TN) <= N;   // block-uniform
        bool used_async = false;

        // ---- A tile: one row (64B) per thread ----
        if (fullk) {
            const unsigned long long ga =
                (unsigned long long)(const void*)(A + (size_t)(m0 + tid) * K + k0);
            const unsigned la = (unsigned)(size_t)(void*)&As[buf][tid * TK];
            asm volatile("global_load_async_to_lds_b128 %0, %1, off"
                         :: "v"(la),      "v"(ga)      : "memory");
            asm volatile("global_load_async_to_lds_b128 %0, %1, off"
                         :: "v"(la + 16), "v"(ga + 16) : "memory");
            asm volatile("global_load_async_to_lds_b128 %0, %1, off"
                         :: "v"(la + 32), "v"(ga + 32) : "memory");
            asm volatile("global_load_async_to_lds_b128 %0, %1, off"
                         :: "v"(la + 48), "v"(ga + 48) : "memory");
            used_async = true;
        } else {
            for (int e = 0; e < TK; ++e) {
                const int kk = k0 + e;
                As[buf][tid * TK + e] =
                    (kk < K) ? A[(size_t)(m0 + tid) * K + kk] : (bf16)0.0f;
            }
        }
        // ---- B tile: one 16B chunk per thread ----
        if (fastB) {
            const unsigned long long gb =
                (unsigned long long)(const void*)(W + (size_t)(n0 + brow) * K + k0 + bseg * 8);
            const unsigned lb = (unsigned)(size_t)(void*)&Bs[buf][brow * TK + bseg * 8];
            asm volatile("global_load_async_to_lds_b128 %0, %1, off"
                         :: "v"(lb), "v"(gb) : "memory");
            used_async = true;
        } else {
            const int nr = n0 + brow;
            for (int e = 0; e < 8; ++e) {
                const int kk = k0 + bseg * 8 + e;
                Bs[buf][brow * TK + bseg * 8 + e] =
                    (nr < N && kk < K) ? W[(size_t)nr * K + kk] : (bf16)0.0f;
            }
        }
        return used_async;
    };

    // ---- prologue: stage first tile into buffer 0 ----
    int cur = 0;
    if (stage(0, 0))
        asm volatile("s_wait_asynccnt 0" ::: "memory");
    __syncthreads();

    for (int k0 = 0; k0 < K; k0 += TK) {
        const int nxt = cur ^ 1;

        // ---- issue async staging of the NEXT tile (overlaps WMMAs below) ----
        bool wait_async = false;
        if (k0 + TK < K)
            wait_async = stage(k0 + TK, nxt);

        // ---- fragments from LDS (two 16B ds reads per chunk) + 8 WMMAs ----
        union Frag { v16bf v; bfx8 h[2]; };
        Frag a[4], b[2];
        for (int nt = 0; nt < 2; ++nt) {
            const int r = wn * 32 + nt * 16 + lm;
            b[nt].h[0] = *(bfx8*)&Bs[cur][r * TK + kh * 8];
            b[nt].h[1] = *(bfx8*)&Bs[cur][r * TK + 16 + kh * 8];
        }
        for (int mt = 0; mt < 4; ++mt) {
            const int r = wm * 64 + mt * 16 + lm;
            a[mt].h[0] = *(bfx8*)&As[cur][r * TK + kh * 8];
            a[mt].h[1] = *(bfx8*)&As[cur][r * TK + 16 + kh * 8];
        }
        for (int mt = 0; mt < 4; ++mt)
            for (int nt = 0; nt < 2; ++nt)
                acc[mt][nt] = __builtin_amdgcn_wmma_f32_16x16x32_bf16(
                    false, a[mt].v, false, b[nt].v,
                    (short)0, acc[mt][nt], false, false);

        // ---- publish next tile: drain ASYNCcnt, then workgroup barrier ----
        if (wait_async)
            asm volatile("s_wait_asynccnt 0" ::: "memory");
        __syncthreads();
        cur = nxt;
    }

    // ---- epilogue: bias + optional ReLU, write fp32 and/or bf16 ----
    for (int mt = 0; mt < 4; ++mt) {
        for (int nt = 0; nt < 2; ++nt) {
            const int n = n0 + wn * 32 + nt * 16 + lm;
            if (n >= N) continue;
            const float bv = bias ? bias[n] : 0.0f;
            for (int g = 0; g < 8; ++g) {
                const int m = m0 + wm * 64 + mt * 16 + g + 8 * kh;
                float v = acc[mt][nt][g] + bv;
                if (relu) v = fmaxf(v, 0.0f);
                const size_t o = (size_t)m * N + n;
                if (Cf) Cf[o] = v;
                if (Cb) Cb[o] = (bf16)v;
            }
        }
    }
}

// ===========================================================================
// fp32 -> bf16 conversion (weights; stays resident in 192MB L2)
// ===========================================================================
__global__ __launch_bounds__(256)
void f32_to_bf16_kernel(const float* __restrict__ src, bf16* __restrict__ dst, int n)
{
    const int i = blockIdx.x * blockDim.x + threadIdx.x;
    if (i < n) dst[i] = (bf16)src[i];
}

// ===========================================================================
// Embed: nodes[b,f, l*8+j] = relu(X[b,f,l] * R_u[f*8+j])  -> bf16
// (src transpose + repeat_interleave + relu + graph-node reshape, fused)
// ===========================================================================
__global__ __launch_bounds__(256)
void embed_kernel(const float* __restrict__ X, const float* __restrict__ Ru,
                  bf16* __restrict__ nodesb, int total)
{
    const int idx = blockIdx.x * blockDim.x + threadIdx.x;
    if (idx >= total) return;
    const int c  = idx % CC;        // l*8 + j
    const int bf = idx / CC;        // b*F_ + f
    const int f  = bf % F_;
    const int l  = c >> 3;
    const int j  = c & 7;
    const float v = X[(size_t)bf * L_ + l] * Ru[f * DOB + j];
    nodesb[idx] = (bf16)fmaxf(v, 0.0f);
}

// ===========================================================================
// Build encoder input x0[L*B, 1040]: cols 0..1023 from x2[b,f, l*8+j],
// cols 1024..1039 = positional encoding sin/cos(t / 256^(i/7)).
// Writes fp32 (residual path) + bf16 (GEMM input).
// ===========================================================================
__global__ __launch_bounds__(256)
void concat_pe_kernel(const float* __restrict__ x2, const float* __restrict__ times,
                      float* __restrict__ x0f, bf16* __restrict__ x0b, int total)
{
    const int idx = blockIdx.x * blockDim.x + threadIdx.x;
    if (idx >= total) return;
    const int col = idx % DD;
    const int row = idx / DD;       // l*B_ + b
    const int l   = row / B_;
    const int b   = row % B_;
    float v;
    if (col < F_ * DOB) {
        const int f = col >> 3, j = col & 7;
        v = x2[((size_t)(b * F_ + f)) * CC + l * 8 + j];
    } else {
        const int i  = col - F_ * DOB;          // 0..15
        const int k  = i & 7;                   // timescale index
        const float ts = powf((float)L_, (float)k * (1.0f / 7.0f));
        const float sc = times[(size_t)b * L_ + l] / ts;
        v = (i < 8) ? sinf(sc) : cosf(sc);
    }
    x0f[idx] = v;
    x0b[idx] = (bf16)v;
}

// ===========================================================================
// Attention (hd=65, not WMMA-shaped; ~5% of FLOPs -> VALU + LDS softmax).
// One block per (query l, head h, batch b). qkv rows indexed by m = l*B + b.
// Writes head output directly as bf16 (input of the Wo GEMM).
// ===========================================================================
__global__ __launch_bounds__(256)
void attn_kernel(const float* __restrict__ qkv, const int* __restrict__ lengths,
                 bf16* __restrict__ attnb)
{
    const int l = blockIdx.x, h = blockIdx.y, b = blockIdx.z;
    const int t = threadIdx.x;      // key index (L_ == 256 == blockDim.x)

    __shared__ float q[72];
    __shared__ float red[256];
    __shared__ float attn[256];

    const size_t qrow = ((size_t)l * B_ + b) * QKVW + (size_t)h * HD;
    if (t < HD) q[t] = qkv[qrow + t];
    __syncthreads();

    const int len = lengths[b];
    const float* krow = qkv + ((size_t)t * B_ + b) * QKVW + DD + (size_t)h * HD;
    float s = 0.0f;
    for (int d = 0; d < HD; ++d) s += q[d] * krow[d];
    s *= 0.124034734589f;                 // 1/sqrt(65)
    if (t >= len) s = -1e9f;              // pad mask

    // block max
    red[t] = s; __syncthreads();
    for (int off = 128; off > 0; off >>= 1) {
        if (t < off) red[t] = fmaxf(red[t], red[t + off]);
        __syncthreads();
    }
    const float mx = red[0]; __syncthreads();

    const float e = expf(s - mx);
    attn[t] = e; red[t] = e; __syncthreads();
    for (int off = 128; off > 0; off >>= 1) {
        if (t < off) red[t] += red[t + off];
        __syncthreads();
    }
    const float denom = red[0]; __syncthreads();

    if (t < HD) {
        float o = 0.0f;
        for (int k = 0; k < L_; ++k)
            o += attn[k] * qkv[((size_t)k * B_ + b) * QKVW + 2 * DD + (size_t)h * HD + t];
        attnb[((size_t)l * B_ + b) * DD + (size_t)h * HD + t] = (bf16)(o / denom);
    }
}

// ===========================================================================
// Residual + LayerNorm: z = LN(resid + delta) * gamma + beta
// One block (256 threads) per row of D=1040; writes fp32 + bf16.
// ===========================================================================
__global__ __launch_bounds__(256)
void ln_kernel(const float* __restrict__ resid, const float* __restrict__ delta,
               const float* __restrict__ gamma, const float* __restrict__ beta,
               float* __restrict__ outf, bf16* __restrict__ outb)
{
    const int row = blockIdx.x;
    const int t   = threadIdx.x;
    __shared__ float rsum[256];
    __shared__ float rsq[256];

    const float* xr = resid + (size_t)row * DD;
    const float* dl = delta + (size_t)row * DD;

    float s = 0.0f, sq = 0.0f;
    for (int c = t; c < DD; c += 256) {
        const float v = xr[c] + dl[c];
        s += v; sq += v * v;
    }
    rsum[t] = s; rsq[t] = sq; __syncthreads();
    for (int off = 128; off > 0; off >>= 1) {
        if (t < off) { rsum[t] += rsum[t + off]; rsq[t] += rsq[t + off]; }
        __syncthreads();
    }
    const float mean = rsum[0] * (1.0f / DD);
    const float var  = rsq[0] * (1.0f / DD) - mean * mean;
    const float inv  = rsqrtf(var + LN_EPS);

    for (int c = t; c < DD; c += 256) {
        const float v = xr[c] + dl[c];
        const float z = (v - mean) * inv * gamma[c] + beta[c];
        outf[(size_t)row * DD + c] = z;
        outb[(size_t)row * DD + c] = (bf16)z;
    }
}

// ===========================================================================
// Pad mask as 0/1 floats into the tail of d_out: mask[b,l] = (l >= len[b])
// ===========================================================================
__global__ __launch_bounds__(256)
void mask_kernel(const int* __restrict__ lengths, float* __restrict__ out)
{
    const int idx = blockIdx.x * blockDim.x + threadIdx.x;
    if (idx >= B_ * L_) return;
    const int b = idx / L_, l = idx % L_;
    out[idx] = (l >= lengths[b]) ? 1.0f : 0.0f;
}

// ===========================================================================
// Host orchestration
// ===========================================================================
static inline void launch_gemm(const bf16* A, const bf16* W, const float* bias,
                               float* Cf, bf16* Cb, int M, int N, int K, int relu,
                               hipStream_t stream)
{
    dim3 grid((N + TN - 1) / TN, M / TM);
    hipLaunchKernelGGL(gemm_bf16_kernel, grid, dim3(256), 0, stream,
                       A, W, bias, Cf, Cb, M, N, K, relu);
}

static inline void launch_cvt(const float* s, bf16* d, int n, hipStream_t stream)
{
    hipLaunchKernelGGL(f32_to_bf16_kernel, dim3((n + 255) / 256), dim3(256), 0, stream,
                       s, d, n);
}

extern "C" void kernel_launch(void* const* d_in, const int* in_sizes, int n_in,
                              void* d_out, int out_size, void* d_ws, size_t ws_size,
                              hipStream_t stream)
{
    // ---- inputs (setup_inputs order) ----
    const float* X     = (const float*)d_in[0];
    const float* times = (const float*)d_in[1];
    const int*   lens  = (const int*)  d_in[2];
    const float* Ru    = (const float*)d_in[3];
    const float* Wv1   = (const float*)d_in[4];
    const float* bv1   = (const float*)d_in[5];
    const float* Wv2   = (const float*)d_in[6];
    const float* bv2   = (const float*)d_in[7];
    const float* Wqkv  = (const float*)d_in[8];
    const float* bqkv  = (const float*)d_in[9];
    const float* Wo    = (const float*)d_in[10];
    const float* bo    = (const float*)d_in[11];
    const float* ln1s  = (const float*)d_in[12];
    const float* ln1b  = (const float*)d_in[13];
    const float* ln2s  = (const float*)d_in[14];
    const float* ln2b  = (const float*)d_in[15];
    const float* Wf1   = (const float*)d_in[16];
    const float* bf1   = (const float*)d_in[17];
    const float* Wf2   = (const float*)d_in[18];
    const float* bf2   = (const float*)d_in[19];

    // ---- workspace bump allocator (lifetimes allow aliasing noted below) ----
    char* p = (char*)d_ws;
    auto alloc = [&](size_t bytes) -> void* {
        void* r = (void*)p;
        p += (bytes + 255) & ~(size_t)255;
        return r;
    };
    // bf16 weight copies (~51 MB total; L2-resident)
    bf16* Wv1b  = (bf16*)alloc((size_t)CC * CC * 2);
    bf16* Wv2b  = (bf16*)alloc((size_t)CC * CC * 2);
    bf16* Wqkvb = (bf16*)alloc((size_t)NL * QKVW * DD * 2);
    bf16* Wob   = (bf16*)alloc((size_t)NL * DD * DD * 2);
    bf16* Wf1b  = (bf16*)alloc((size_t)NL * DFFN * DD * 2);
    bf16* Wf2b  = (bf16*)alloc((size_t)NL * DD * DFFN * 2);
    // activations
    bf16*  nodesb = (bf16*)alloc((size_t)MBF * CC * 2);   // also reused as attnb
    bf16*  x1b    = (bf16*)alloc((size_t)MBF * CC * 2);
    float* qkvf   = (float*)alloc((size_t)MLB * QKVW * 4);
    float* x2f    = qkvf;                                  // alias: dead before qkv GEMM
    float* x0f    = (float*)alloc((size_t)MLB * DD * 4);
    bf16*  x0b    = (bf16*)alloc((size_t)MLB * DD * 2);
    bf16*  attnb  = nodesb;                                // alias: nodes dead after GEMM1
    float* aprojf = (float*)alloc((size_t)MLB * DD * 4);
    float* x1f    = (float*)alloc((size_t)MLB * DD * 4);
    bf16*  x1benc = (bf16*)alloc((size_t)MLB * DD * 2);
    bf16*  hb     = (bf16*)alloc((size_t)MLB * DFFN * 2);
    float* fff    = (float*)alloc((size_t)MLB * DD * 4);
    (void)ws_size; (void)in_sizes; (void)n_in; (void)out_size;

    // ---- 1) weights -> bf16 ----
    launch_cvt(Wv1,  Wv1b,  CC * CC,            stream);
    launch_cvt(Wv2,  Wv2b,  CC * CC,            stream);
    launch_cvt(Wqkv, Wqkvb, NL * QKVW * DD,     stream);
    launch_cvt(Wo,   Wob,   NL * DD * DD,       stream);
    launch_cvt(Wf1,  Wf1b,  NL * DFFN * DD,     stream);
    launch_cvt(Wf2,  Wf2b,  NL * DD * DFFN,     stream);

    // ---- 2) embed: relu(repeat(X) * R_u) -> nodes bf16 [8192, 2048] ----
    {
        const int total = MBF * CC;
        hipLaunchKernelGGL(embed_kernel, dim3((total + 255) / 256), dim3(256), 0,
                           stream, X, Ru, nodesb, total);
    }

    // ---- 3) ObsProp (softmax(ones) collapses to identity coef):
    //         x1 = relu(nodes @ Wv1^T + bv1); x2 = relu(x1 @ Wv2^T + bv2) ----
    launch_gemm(nodesb, Wv1b, bv1, nullptr, x1b, MBF, CC, CC, 1, stream);
    launch_gemm(x1b,    Wv2b, bv2, x2f,  nullptr, MBF, CC, CC, 1, stream);

    // ---- 4) reshape + concat positional encoding -> x0 [16384, 1040] ----
    {
        const int total = MLB * DD;
        hipLaunchKernelGGL(concat_pe_kernel, dim3((total + 255) / 256), dim3(256), 0,
                           stream, x2f, times, x0f, x0b, total);
    }

    // ---- 5) transformer encoder, 2 post-LN layers ----
    for (int i = 0; i < NL; ++i) {
        // qkv = x0 @ Wqkv_i^T + bqkv_i            [16384, 3120]
        launch_gemm(x0b, Wqkvb + (size_t)i * QKVW * DD, bqkv + (size_t)i * QKVW,
                    qkvf, nullptr, MLB, QKVW, DD, 0, stream);
        // per-head attention (VALU + LDS softmax) -> bf16 [16384, 1040]
        hipLaunchKernelGGL(attn_kernel, dim3(L_, NH, B_), dim3(256), 0, stream,
                           qkvf, lens, attnb);
        // out-proj
        launch_gemm(attnb, Wob + (size_t)i * DD * DD, bo + (size_t)i * DD,
                    aprojf, nullptr, MLB, DD, DD, 0, stream);
        // x1 = LN(x0 + aproj)
        hipLaunchKernelGGL(ln_kernel, dim3(MLB), dim3(256), 0, stream,
                           x0f, aprojf, ln1s + (size_t)i * DD, ln1b + (size_t)i * DD,
                           x1f, x1benc);
        // h = relu(x1 @ Wf1_i^T + bf1_i)          [16384, 2048]
        launch_gemm(x1benc, Wf1b + (size_t)i * DFFN * DD, bf1 + (size_t)i * DFFN,
                    nullptr, hb, MLB, DFFN, DD, 1, stream);
        // ffn = h @ Wf2_i^T + bf2_i
        launch_gemm(hb, Wf2b + (size_t)i * DD * DFFN, bf2 + (size_t)i * DD,
                    fff, nullptr, MLB, DD, DFFN, 0, stream);
        // x0 = LN(x1 + ffn)   (feeds next layer / final output)
        hipLaunchKernelGGL(ln_kernel, dim3(MLB), dim3(256), 0, stream,
                           x1f, fff, ln2s + (size_t)i * DD, ln2b + (size_t)i * DD,
                           x0f, x0b);
    }

    // ---- 6) outputs: [L,B,D] fp32 then mask [B,L] as 0/1 floats ----
    hipMemcpyAsync(d_out, x0f, (size_t)MLB * DD * sizeof(float),
                   hipMemcpyDeviceToDevice, stream);
    hipLaunchKernelGGL(mask_kernel, dim3((B_ * L_ + 255) / 256), dim3(256), 0, stream,
                       lens, (float*)d_out + (size_t)MLB * DD);
}
